// Net_7155415515574
// MI455X (gfx1250) — matrix operand order repaired and checked by reading, not compiled, not run on hardware
//
#include <hip/hip_runtime.h>
#include <stdint.h>

#define DIM 64
#define NN 32768
#define EE 65536
#define GG 2048
#define TT 65536

typedef __attribute__((ext_vector_type(16))) __bf16 v16bf;
typedef __attribute__((ext_vector_type(8))) float v8f;

union BF16x16 { v16bf v; unsigned short u[16]; };

__device__ __forceinline__ unsigned short f2bf(float f) {
  union { float f; unsigned u; } x; x.f = f;
  unsigned r = 0x7FFFu + ((x.u >> 16) & 1u);
  return (unsigned short)((x.u + r) >> 16);
}
__device__ __forceinline__ float bf2f(unsigned short b) {
  union { unsigned u; float f; } x; x.u = ((unsigned)b) << 16; return x.f;
}
__device__ __forceinline__ float sigmoidf(float x){ return 1.0f/(1.0f+__expf(-x)); }

__device__ __forceinline__ float wave_sum32(float v){
  #pragma unroll
  for (int m = 16; m >= 1; m >>= 1) v += __shfl_xor(v, m, 32);
  return v;
}
// reduce within each 16-lane half of a wave32 (masks < 16 keep halves separate)
__device__ __forceinline__ float half_sum16(float v){
  #pragma unroll
  for (int m = 8; m >= 1; m >>= 1) v += __shfl_xor(v, m, 32);
  return v;
}

__device__ __forceinline__ unsigned fkey(float f){
  unsigned u = __float_as_uint(f);
  return (u & 0x80000000u) ? ~u : (u | 0x80000000u);
}
__device__ __forceinline__ float funkey(unsigned k){
  unsigned u = (k & 0x80000000u) ? (k & 0x7FFFFFFFu) : ~k;
  return __uint_as_float(u);
}

// ---------------- preprocess MLP: Linear(16,64)+LN+ReLU, Linear(64,64)+LN+ReLU
__global__ void __launch_bounds__(128)
k_pre(const float* __restrict__ x,
      const float* __restrict__ w1, const float* __restrict__ b1,
      const float* __restrict__ g1, const float* __restrict__ be1,
      const float* __restrict__ w2, const float* __restrict__ b2,
      const float* __restrict__ g2, const float* __restrict__ be2,
      float* __restrict__ out) {
  __shared__ float hbuf[4][DIM];
  int lane = threadIdx.x & 31, w = threadIdx.x >> 5;
  int n = blockIdx.x * 4 + w;
  int o0 = lane, o1 = lane + 32;
  float a0 = b1[o0], a1 = b1[o1];
  const float* xr = x + (size_t)n * 16;
  #pragma unroll
  for (int k = 0; k < 16; ++k) {
    float xv = xr[k];
    a0 = fmaf(xv, w1[o0*16+k], a0);
    a1 = fmaf(xv, w1[o1*16+k], a1);
  }
  float s  = wave_sum32(a0 + a1);
  float ss = wave_sum32(a0*a0 + a1*a1);
  float mean = s * (1.0f/64.0f);
  float var  = ss * (1.0f/64.0f) - mean*mean;
  float rs   = rsqrtf(var + 1e-5f);
  float v0 = fmaxf((a0-mean)*rs*g1[o0] + be1[o0], 0.0f);
  float v1 = fmaxf((a1-mean)*rs*g1[o1] + be1[o1], 0.0f);
  hbuf[w][o0] = v0; hbuf[w][o1] = v1;
  __syncthreads();
  a0 = b2[o0]; a1 = b2[o1];
  #pragma unroll 8
  for (int j = 0; j < DIM; ++j) {
    float hv = hbuf[w][j];
    a0 = fmaf(hv, w2[o0*DIM+j], a0);
    a1 = fmaf(hv, w2[o1*DIM+j], a1);
  }
  s = wave_sum32(a0+a1); ss = wave_sum32(a0*a0+a1*a1);
  mean = s*(1.0f/64.0f); var = ss*(1.0f/64.0f)-mean*mean; rs = rsqrtf(var+1e-5f);
  v0 = fmaxf((a0-mean)*rs*g2[o0]+be2[o0], 0.0f);
  v1 = fmaxf((a1-mean)*rs*g2[o1]+be2[o1], 0.0f);
  out[(size_t)n*DIM+o0] = v0;
  out[(size_t)n*DIM+o1] = v1;
}

// ---------------- edge encoder layer 1: Linear(4,64)+LN+ReLU -> bf16
__global__ void __launch_bounds__(128)
k_enc1(const float* __restrict__ ea, const float* __restrict__ w1,
       const float* __restrict__ b1, const float* __restrict__ g1,
       const float* __restrict__ be1, unsigned short* __restrict__ ebf) {
  int lane = threadIdx.x & 31, w = threadIdx.x >> 5;
  int e = blockIdx.x * 4 + w;
  int o0 = lane, o1 = lane + 32;
  const float* er = ea + (size_t)e * 4;
  float e0 = er[0], e1 = er[1], e2 = er[2], e3 = er[3];
  float a0 = b1[o0] + e0*w1[o0*4+0] + e1*w1[o0*4+1] + e2*w1[o0*4+2] + e3*w1[o0*4+3];
  float a1 = b1[o1] + e0*w1[o1*4+0] + e1*w1[o1*4+1] + e2*w1[o1*4+2] + e3*w1[o1*4+3];
  float s  = wave_sum32(a0 + a1);
  float ss = wave_sum32(a0*a0 + a1*a1);
  float mean = s * (1.0f/64.0f);
  float var  = ss * (1.0f/64.0f) - mean*mean;
  float rs   = rsqrtf(var + 1e-5f);
  float v0 = fmaxf((a0-mean)*rs*g1[o0] + be1[o0], 0.0f);
  float v1 = fmaxf((a1-mean)*rs*g1[o1] + be1[o1], 0.0f);
  ebf[(size_t)e*DIM+o0] = f2bf(v0);
  ebf[(size_t)e*DIM+o1] = f2bf(v1);
}

// ---------------- f32 -> bf16 convert
__global__ void k_cvt_bf16(const float* __restrict__ src, unsigned short* __restrict__ dst, int n) {
  int i = blockIdx.x * 256 + threadIdx.x;
  if (i < n) dst[i] = f2bf(src[i]);
}

// ---------------- in-degree count
__global__ void k_count(const int* __restrict__ ei, float* __restrict__ cnt) {
  int e = blockIdx.x * 256 + threadIdx.x;
  atomicAdd(&cnt[ei[EE + e]], 1.0f);
}

// ---------------- big WMMA GEMM: z = e_bf[E,64] @ W2_bf^T + b2 (pre-LN, bf16)
// block = 8 waves, owns one 16-edge M-tile across all 4096 cols -> exact LN stats.
// z is streamed with non-temporal hints (512MB >> 192MB L2).
__global__ void __launch_bounds__(256)
k_gemm_enc(const unsigned short* __restrict__ ebf,
           const unsigned short* __restrict__ w2bf,
           const float* __restrict__ b2,
           unsigned short* __restrict__ z,
           float* __restrict__ mbuf, float* __restrict__ rsbuf) {
  __shared__ float ssum[16], ssq[16];
  if (threadIdx.x < 16) { ssum[threadIdx.x] = 0.f; ssq[threadIdx.x] = 0.f; }
  __syncthreads();
  int lane = threadIdx.x & 31, w = threadIdx.x >> 5;
  int hi = lane >> 4, l15 = lane & 15;
  int m0 = blockIdx.x * 16;
  // A fragments (16x32 bf16 layout): lane<16 holds K {0..7,16..23}, lane>=16 holds K {8..15,24..31}
  const unsigned short* Ar = ebf + (size_t)(m0 + l15) * 64;
  BF16x16 A0, A1;
  #pragma unroll
  for (int j = 0; j < 8; ++j) {
    A0.u[j]   = Ar[hi*8 + j];
    A0.u[j+8] = Ar[16 + hi*8 + j];
    A1.u[j]   = Ar[32 + hi*8 + j];
    A1.u[j+8] = Ar[48 + hi*8 + j];
  }
  float psum[8], psq[8];
  #pragma unroll
  for (int r = 0; r < 8; ++r) { psum[r] = 0.f; psq[r] = 0.f; }
  for (int t = 0; t < 32; ++t) {
    int n0 = (w*32 + t) * 16;
    // B (32x16): lane = column, lanes 0-15 hold K {0..7,16..23}, lanes 16-31 hold K {8..15,24..31}
    const unsigned short* Br = w2bf + (size_t)(n0 + l15) * 64;
    BF16x16 B0, B1;
    #pragma unroll
    for (int j = 0; j < 8; ++j) { B0.u[j] = Br[hi*8 + j]; B0.u[j+8] = Br[16 + hi*8 + j]; B1.u[j] = Br[32 + hi*8 + j]; B1.u[j+8] = Br[48 + hi*8 + j]; }
    // independent accumulators: no WMMA->WMMA RAW inside the iteration
    v8f c0 = {}, c1 = {};
    c0 = __builtin_amdgcn_wmma_f32_16x16x32_bf16(false, A0.v, false, B0.v, (short)0, c0, false, false);
    c1 = __builtin_amdgcn_wmma_f32_16x16x32_bf16(false, A1.v, false, B1.v, (short)0, c1, false, false);
    float bias = b2[n0 + l15];
    size_t base = (size_t)(m0 + hi*8) * 4096 + n0 + l15;
    #pragma unroll
    for (int r = 0; r < 8; ++r) {
      float v = c0[r] + c1[r] + bias;
      __builtin_nontemporal_store(f2bf(v), &z[base + (size_t)r * 4096]);
      psum[r] += v; psq[r] += v*v;
    }
  }
  #pragma unroll
  for (int r = 0; r < 8; ++r) { psum[r] = half_sum16(psum[r]); psq[r] = half_sum16(psq[r]); }
  if (l15 == 0) {
    #pragma unroll
    for (int r = 0; r < 8; ++r) {
      atomicAdd(&ssum[hi*8 + r], psum[r]);
      atomicAdd(&ssq [hi*8 + r], psq[r]);
    }
  }
  __syncthreads();
  if (threadIdx.x < 16) {
    float mean = ssum[threadIdx.x] * (1.0f/4096.0f);
    float var  = ssq [threadIdx.x] * (1.0f/4096.0f) - mean*mean;
    mbuf [m0 + threadIdx.x] = mean;
    rsbuf[m0 + threadIdx.x] = rsqrtf(var + 1e-5f);
  }
}

// ---------------- NNConv message with LN folded: one wave per edge
// msg_o = rs*(sum_i a_i g_io z_io - m * sum_i a_i g_io) + sum_i a_i be_io
__global__ void __launch_bounds__(256)
k_msg(const float* __restrict__ out, const int* __restrict__ ei,
      const unsigned short* __restrict__ z,
      const float* __restrict__ mbuf, const float* __restrict__ rsbuf,
      const float* __restrict__ g, const float* __restrict__ be,
      float* __restrict__ accum) {
  __shared__ float abuf[8][DIM];
  int lane = threadIdx.x & 31, w = threadIdx.x >> 5;
  int e = blockIdx.x * 8 + w;
  int src = ei[e], dst = ei[EE + e];
  abuf[w][lane]      = out[(size_t)src*DIM + lane];
  abuf[w][lane + 32] = out[(size_t)src*DIM + lane + 32];
  __syncthreads();
  float S1a=0,S1b=0,S2a=0,S2b=0,S3a=0,S3b=0;
  const unsigned short* zr = z + (size_t)e * 4096;
  int oc = 2*lane;
  for (int i = 0; i < DIM; ++i) {
    float ai = abuf[w][i];
    int off = i*64 + oc;
    unsigned zz = __builtin_nontemporal_load((const unsigned*)(zr + off));
    float z0 = bf2f((unsigned short)(zz & 0xffffu));
    float z1 = bf2f((unsigned short)(zz >> 16));
    float g0 = g[off], g1v = g[off+1];
    float b0 = be[off], b1v = be[off+1];
    S1a = fmaf(ai*g0,  z0, S1a); S1b = fmaf(ai*g1v, z1, S1b);
    S2a = fmaf(ai, g0,  S2a);    S2b = fmaf(ai, g1v, S2b);
    S3a = fmaf(ai, b0,  S3a);    S3b = fmaf(ai, b1v, S3b);
  }
  float mm = mbuf[e], rr = rsbuf[e];
  float m0v = rr*(S1a - mm*S2a) + S3a;
  float m1v = rr*(S1b - mm*S2b) + S3b;
  atomicAdd(&accum[(size_t)dst*DIM + oc],     m0v);
  atomicAdd(&accum[(size_t)dst*DIM + oc + 1], m1v);
}

// ---------------- scatter-mean epilogue + GRU cell, one wave per node, in-place
__global__ void __launch_bounds__(256)
k_gru(float* __restrict__ out, const float* __restrict__ accum,
      const float* __restrict__ cnt, const float* __restrict__ convb,
      const float* __restrict__ wih, const float* __restrict__ whh,
      const float* __restrict__ bih, const float* __restrict__ bhh) {
  __shared__ float mb[8][DIM], hb[8][DIM];
  int lane = threadIdx.x & 31, w = threadIdx.x >> 5;
  int n = blockIdx.x * 8 + w;
  float inv = 1.0f / fmaxf(cnt[n], 1.0f);
  int o0 = lane, o1 = lane + 32;
  mb[w][o0] = fmaxf(accum[(size_t)n*DIM+o0]*inv + convb[o0], 0.0f);
  mb[w][o1] = fmaxf(accum[(size_t)n*DIM+o1]*inv + convb[o1], 0.0f);
  hb[w][o0] = out[(size_t)n*DIM+o0];
  hb[w][o1] = out[(size_t)n*DIM+o1];
  __syncthreads();
  float gi[6], gh[6];
  #pragma unroll
  for (int q = 0; q < 6; ++q) {
    int ro = (q>>1)*64 + (q&1)*32 + lane;
    float ai = bih[ro], ah = bhh[ro];
    const float* wi = wih + (size_t)ro*DIM;
    const float* wh = whh + (size_t)ro*DIM;
    #pragma unroll 8
    for (int j = 0; j < DIM; ++j) {
      ai = fmaf(mb[w][j], wi[j], ai);
      ah = fmaf(hb[w][j], wh[j], ah);
    }
    gi[q] = ai; gh[q] = ah;
  }
  float r0 = sigmoidf(gi[0]+gh[0]);
  float z0 = sigmoidf(gi[2]+gh[2]);
  float n0 = tanhf(gi[4] + r0*gh[4]);
  float hn0 = (1.0f - z0)*n0 + z0*hb[w][o0];
  float r1 = sigmoidf(gi[1]+gh[1]);
  float z1 = sigmoidf(gi[3]+gh[3]);
  float n1 = tanhf(gi[5] + r1*gh[5]);
  float hn1 = (1.0f - z1)*n1 + z1*hb[w][o1];
  out[(size_t)n*DIM+o0] = hn0;
  out[(size_t)n*DIM+o1] = hn1;
}

// ---------------- Set2Set LSTM cell, one wave per group
__global__ void __launch_bounds__(256)
k_lstm(const float* __restrict__ qstar, float* __restrict__ qh, float* __restrict__ qc,
       const float* __restrict__ wih, const float* __restrict__ whh,
       const float* __restrict__ bih, const float* __restrict__ bhh) {
  __shared__ float qs[8][128], qhb[8][DIM];
  int lane = threadIdx.x & 31, w = threadIdx.x >> 5;
  int g = blockIdx.x * 8 + w;
  #pragma unroll
  for (int j = 0; j < 4; ++j) qs[w][lane + 32*j] = qstar[(size_t)g*128 + lane + 32*j];
  qhb[w][lane]      = qh[(size_t)g*DIM + lane];
  qhb[w][lane + 32] = qh[(size_t)g*DIM + lane + 32];
  __syncthreads();
  float gate[8];
  #pragma unroll
  for (int k = 0; k < 8; ++k) {
    int ro = lane + 32*k;
    float a = bih[ro] + bhh[ro];
    const float* wi = wih + (size_t)ro*128;
    const float* wh = whh + (size_t)ro*DIM;
    #pragma unroll 8
    for (int j = 0; j < 128; ++j) a = fmaf(qs[w][j], wi[j], a);
    #pragma unroll 8
    for (int j = 0; j < DIM; ++j) a = fmaf(qhb[w][j], wh[j], a);
    gate[k] = a;
  }
  int d0 = lane, d1 = lane + 32;
  float c0 = sigmoidf(gate[2])*qc[(size_t)g*DIM+d0] + sigmoidf(gate[0])*tanhf(gate[4]);
  float c1 = sigmoidf(gate[3])*qc[(size_t)g*DIM+d1] + sigmoidf(gate[1])*tanhf(gate[5]);
  float h0 = sigmoidf(gate[6])*tanhf(c0);
  float h1 = sigmoidf(gate[7])*tanhf(c1);
  qc[(size_t)g*DIM+d0] = c0; qc[(size_t)g*DIM+d1] = c1;
  qh[(size_t)g*DIM+d0] = h0; qh[(size_t)g*DIM+d1] = h1;
}

__global__ void k_escore(const float* __restrict__ out, const float* __restrict__ qh,
                         const int* __restrict__ nb, float* __restrict__ escore,
                         unsigned* __restrict__ emaxk) {
  int n = blockIdx.x * 256 + threadIdx.x;
  int b = nb[n];
  const float* orow = out + (size_t)n*DIM;
  const float* qrow = qh + (size_t)b*DIM;
  float s = 0.f;
  #pragma unroll 8
  for (int d = 0; d < DIM; ++d) s = fmaf(orow[d], qrow[d], s);
  escore[n] = s;
  atomicMax(&emaxk[b], fkey(s));
}

__global__ void k_aexp(const float* __restrict__ escore, const unsigned* __restrict__ emaxk,
                       const int* __restrict__ nb, float* __restrict__ avec,
                       float* __restrict__ asum) {
  int n = blockIdx.x * 256 + threadIdx.x;
  int b = nb[n];
  float a = __expf(escore[n] - funkey(emaxk[b]));
  avec[n] = a;
  atomicAdd(&asum[b], a);
}

__global__ void __launch_bounds__(256)
k_rvec(const float* __restrict__ out, const float* __restrict__ avec,
       const float* __restrict__ asum, const int* __restrict__ nb,
       float* __restrict__ rvec) {
  int lane = threadIdx.x & 31, w = threadIdx.x >> 5;
  int n = blockIdx.x * 8 + w;
  int b = nb[n];
  float coef = avec[n] / asum[b];
  atomicAdd(&rvec[(size_t)b*DIM + lane],      coef*out[(size_t)n*DIM + lane]);
  atomicAdd(&rvec[(size_t)b*DIM + lane + 32], coef*out[(size_t)n*DIM + lane + 32]);
}

__global__ void k_qstar(const float* __restrict__ qh, const float* __restrict__ rvec,
                        float* __restrict__ qstar) {
  int i = blockIdx.x * 256 + threadIdx.x;   // over G*128
  int g = i >> 7, j = i & 127;
  qstar[i] = (j < DIM) ? qh[(size_t)g*DIM + j] : rvec[(size_t)g*DIM + j - DIM];
}

// ---------------- build predict input P[T,256] bf16
__global__ void __launch_bounds__(256)
k_gatherP(const float* __restrict__ out, const float* __restrict__ qstar,
          const int* __restrict__ tix, const int* __restrict__ tbi,
          unsigned short* __restrict__ P) {
  int lane = threadIdx.x & 31, w = threadIdx.x >> 5;
  int t = blockIdx.x * 8 + w;
  int t0 = tix[t], t1 = tix[TT + t], b = tbi[t];
  unsigned short* pr = P + (size_t)t * 256;
  #pragma unroll
  for (int jj = 0; jj < 8; ++jj) {
    int j = lane*8 + jj;
    float v;
    if (j < 64)       v = out[(size_t)t0*DIM + j];
    else if (j < 128) v = out[(size_t)t1*DIM + (j - 64)];
    else              v = qstar[(size_t)b*128 + (j - 128)];
    pr[j] = f2bf(v);
  }
}

// ---------------- predict MLP layer: Y = relu(LN(X@W^T + b)), WMMA bf16, K=256
// K unrolled by 64 with 4 independent accumulators (2 N-tiles x 2 K-streams).
__global__ void __launch_bounds__(256)
k_mlp(const unsigned short* __restrict__ X, const unsigned short* __restrict__ Wb,
      const float* __restrict__ bias, const float* __restrict__ gam,
      const float* __restrict__ bet, unsigned short* __restrict__ Y) {
  __shared__ float ssum[16], ssq[16], smean[16], srstd[16];
  if (threadIdx.x < 16) { ssum[threadIdx.x] = 0.f; ssq[threadIdx.x] = 0.f; }
  __syncthreads();
  int lane = threadIdx.x & 31, w = threadIdx.x >> 5;
  int hi = lane >> 4, l15 = lane & 15;
  int m0 = blockIdx.x * 16;
  int n0a = w*32, n0b = w*32 + 16;
  v8f c0a = {}, c0b = {}, c1a = {}, c1b = {};
  const unsigned short* Ar  = X  + (size_t)(m0 + l15) * 256;
  const unsigned short* Bra = Wb + (size_t)(n0a + l15) * 256;
  const unsigned short* Brb = Wb + (size_t)(n0b + l15) * 256;
  for (int kc = 0; kc < 4; ++kc) {
    int k0 = kc*64;
    BF16x16 A0, A1, B0a, B1a, B0b, B1b;
    #pragma unroll
    for (int j = 0; j < 8; ++j) {
      A0.u[j]   = Ar[k0 + hi*8 + j];
      A0.u[j+8] = Ar[k0 + 16 + hi*8 + j];
      A1.u[j]   = Ar[k0 + 32 + hi*8 + j];
      A1.u[j+8] = Ar[k0 + 48 + hi*8 + j];
    }
    #pragma unroll
    for (int j = 0; j < 8; ++j) {
      B0a.u[j] = Bra[k0 + hi*8 + j];      B0a.u[j+8] = Bra[k0 + 16 + hi*8 + j];
      B0b.u[j] = Bra[k0 + 32 + hi*8 + j]; B0b.u[j+8] = Bra[k0 + 48 + hi*8 + j];
      B1a.u[j] = Brb[k0 + hi*8 + j];      B1a.u[j+8] = Brb[k0 + 16 + hi*8 + j];
      B1b.u[j] = Brb[k0 + 32 + hi*8 + j]; B1b.u[j+8] = Brb[k0 + 48 + hi*8 + j];
    }
    // 4 independent WMMAs per iteration (no intra-iteration RAW)
    c0a = __builtin_amdgcn_wmma_f32_16x16x32_bf16(false, A0.v, false, B0a.v, (short)0, c0a, false, false);
    c1a = __builtin_amdgcn_wmma_f32_16x16x32_bf16(false, A0.v, false, B1a.v, (short)0, c1a, false, false);
    c0b = __builtin_amdgcn_wmma_f32_16x16x32_bf16(false, A1.v, false, B0b.v, (short)0, c0b, false, false);
    c1b = __builtin_amdgcn_wmma_f32_16x16x32_bf16(false, A1.v, false, B1b.v, (short)0, c1b, false, false);
  }
  float ba = bias[n0a + l15], bb = bias[n0b + l15];
  float va[8], vb[8], psum[8], psq[8];
  #pragma unroll
  for (int r = 0; r < 8; ++r) {
    va[r] = c0a[r] + c0b[r] + ba;
    vb[r] = c1a[r] + c1b[r] + bb;
    psum[r] = va[r] + vb[r];
    psq[r]  = va[r]*va[r] + vb[r]*vb[r];
  }
  #pragma unroll
  for (int r = 0; r < 8; ++r) { psum[r] = half_sum16(psum[r]); psq[r] = half_sum16(psq[r]); }
  if (l15 == 0) {
    #pragma unroll
    for (int r = 0; r < 8; ++r) {
      atomicAdd(&ssum[hi*8 + r], psum[r]);
      atomicAdd(&ssq [hi*8 + r], psq[r]);
    }
  }
  __syncthreads();
  if (threadIdx.x < 16) {
    float mean = ssum[threadIdx.x] * (1.0f/256.0f);
    float var  = ssq [threadIdx.x] * (1.0f/256.0f) - mean*mean;
    smean[threadIdx.x] = mean;
    srstd[threadIdx.x] = rsqrtf(var + 1e-5f);
  }
  __syncthreads();
  float g0 = gam[n0a + l15], g1v = gam[n0b + l15];
  float e0 = bet[n0a + l15], e1v = bet[n0b + l15];
  #pragma unroll
  for (int r = 0; r < 8; ++r) {
    int mrow = r + hi*8;
    float mean = smean[mrow], rs = srstd[mrow];
    float y0 = fmaxf((va[r]-mean)*rs*g0 + e0, 0.0f);
    float y1 = fmaxf((vb[r]-mean)*rs*g1v + e1v, 0.0f);
    size_t base = (size_t)(m0 + mrow) * 256;
    Y[base + n0a + l15] = f2bf(y0);
    Y[base + n0b + l15] = f2bf(y1);
  }
}

// ---------------- final: p = Y @ W3^T + b3, gather target class, one wave/target
__global__ void __launch_bounds__(256)
k_final(const unsigned short* __restrict__ Y, const float* __restrict__ w3,
        const float* __restrict__ b3, const int* __restrict__ cls,
        float* __restrict__ outp) {
  int lane = threadIdx.x & 31, w = threadIdx.x >> 5;
  int t = blockIdx.x * 8 + w;
  int c = cls[t];
  const unsigned short* yr = Y + (size_t)t * 256;
  const float* wr = w3 + (size_t)c * 256;
  float s = 0.f;
  #pragma unroll
  for (int jj = 0; jj < 8; ++jj) {
    int j = lane + 32*jj;
    s = fmaf(bf2f(yr[j]), wr[j], s);
  }
  s = wave_sum32(s);
  if (lane == 0) outp[t] = s + b3[c];
}

extern "C" void kernel_launch(void* const* d_in, const int* in_sizes, int n_in,
                              void* d_out, int out_size, void* d_ws, size_t ws_size,
                              hipStream_t stream) {
  (void)in_sizes; (void)n_in; (void)out_size; (void)ws_size;
  const float* x            = (const float*)d_in[0];
  const float* edge_attr    = (const float*)d_in[1];
  const int*   edge_index   = (const int*)  d_in[2];
  const int*   node_batch   = (const int*)  d_in[3];
  const int*   target_index = (const int*)  d_in[4];
  const int*   target_batch = (const int*)  d_in[5];
  const int*   target_class = (const int*)  d_in[6];
  const float* pre_w1 = (const float*)d_in[7];
  const float* pre_b1 = (const float*)d_in[8];
  const float* pre_g1 = (const float*)d_in[9];
  const float* pre_be1= (const float*)d_in[10];
  const float* pre_w2 = (const float*)d_in[11];
  const float* pre_b2 = (const float*)d_in[12];
  const float* pre_g2 = (const float*)d_in[13];
  const float* pre_be2= (const float*)d_in[14];
  const float* enc_w1 = (const float*)d_in[15];
  const float* enc_b1 = (const float*)d_in[16];
  const float* enc_g1 = (const float*)d_in[17];
  const float* enc_be1= (const float*)d_in[18];
  const float* enc_w2 = (const float*)d_in[19];
  const float* enc_b2 = (const float*)d_in[20];
  const float* enc_g2 = (const float*)d_in[21];
  const float* enc_be2= (const float*)d_in[22];
  const float* conv_b = (const float*)d_in[23];
  const float* gru_wih= (const float*)d_in[24];
  const float* gru_whh= (const float*)d_in[25];
  const float* gru_bih= (const float*)d_in[26];
  const float* gru_bhh= (const float*)d_in[27];
  const float* lstm_wih=(const float*)d_in[28];
  const float* lstm_whh=(const float*)d_in[29];
  const float* lstm_bih=(const float*)d_in[30];
  const float* lstm_bhh=(const float*)d_in[31];
  const float* prd_w1 = (const float*)d_in[32];
  const float* prd_b1 = (const float*)d_in[33];
  const float* prd_g1 = (const float*)d_in[34];
  const float* prd_be1= (const float*)d_in[35];
  const float* prd_w2 = (const float*)d_in[36];
  const float* prd_b2 = (const float*)d_in[37];
  const float* prd_g2 = (const float*)d_in[38];
  const float* prd_be2= (const float*)d_in[39];
  const float* prd_w3 = (const float*)d_in[40];
  const float* prd_b3 = (const float*)d_in[41];

  char* base = (char*)d_ws; size_t off = 0;
  auto carve = [&](size_t bytes) -> void* {
    void* r = base + off; off += (bytes + 255) & ~(size_t)255; return r;
  };
  float* out_s  = (float*)carve((size_t)NN*DIM*4);
  float* accum  = (float*)carve((size_t)NN*DIM*4);
  float* cnt    = (float*)carve((size_t)NN*4);
  unsigned short* ebf  = (unsigned short*)carve((size_t)EE*DIM*2);
  unsigned short* w2bf = (unsigned short*)carve((size_t)4096*64*2);
  float* mbuf   = (float*)carve((size_t)EE*4);
  float* rsbuf  = (float*)carve((size_t)EE*4);
  float* qh     = (float*)carve((size_t)GG*DIM*4);
  float* qc     = (float*)carve((size_t)GG*DIM*4);
  float* qstar  = (float*)carve((size_t)GG*128*4);
  unsigned* emaxk = (unsigned*)carve((size_t)GG*4);
  float* asum   = (float*)carve((size_t)GG*4);
  float* rvec   = (float*)carve((size_t)GG*DIM*4);
  float* escore = (float*)carve((size_t)NN*4);
  float* avec   = (float*)carve((size_t)NN*4);
  unsigned short* w1pbf = (unsigned short*)carve((size_t)256*256*2);
  unsigned short* w2pbf = (unsigned short*)carve((size_t)256*256*2);
  unsigned short* P = (unsigned short*)carve((size_t)TT*256*2);
  unsigned short* Q = (unsigned short*)carve((size_t)TT*256*2);
  unsigned short* z = (unsigned short*)carve((size_t)EE*4096*2);   // 512 MB, last

  // weight conversions to bf16 (L2-resident)
  k_cvt_bf16<<<(4096*64+255)/256,256,0,stream>>>(enc_w2, w2bf, 4096*64);
  k_cvt_bf16<<<(256*256+255)/256,256,0,stream>>>(prd_w1, w1pbf, 256*256);
  k_cvt_bf16<<<(256*256+255)/256,256,0,stream>>>(prd_w2, w2pbf, 256*256);

  // node preprocess + edge encoder L1
  k_pre <<<NN/4,128,0,stream>>>(x, pre_w1, pre_b1, pre_g1, pre_be1,
                                pre_w2, pre_b2, pre_g2, pre_be2, out_s);
  k_enc1<<<EE/4,128,0,stream>>>(edge_attr, enc_w1, enc_b1, enc_g1, enc_be1, ebf);

  // big WMMA GEMM: z (pre-LN) + per-edge LN stats
  k_gemm_enc<<<EE/16,256,0,stream>>>(ebf, w2bf, enc_b2, z, mbuf, rsbuf);

  // in-degree counts
  hipMemsetAsync(cnt, 0, (size_t)NN*4, stream);
  k_count<<<EE/256,256,0,stream>>>(edge_index, cnt);

  // 3x (NNConv + GRU)
  for (int it = 0; it < 3; ++it) {
    hipMemsetAsync(accum, 0, (size_t)NN*DIM*4, stream);
    k_msg<<<EE/8,256,0,stream>>>(out_s, edge_index, z, mbuf, rsbuf, enc_g2, enc_be2, accum);
    k_gru<<<NN/8,256,0,stream>>>(out_s, accum, cnt, conv_b, gru_wih, gru_whh, gru_bih, gru_bhh);
  }

  // Set2Set, 3 steps
  hipMemsetAsync(qh,    0, (size_t)GG*DIM*4, stream);
  hipMemsetAsync(qc,    0, (size_t)GG*DIM*4, stream);
  hipMemsetAsync(qstar, 0, (size_t)GG*128*4, stream);
  for (int it = 0; it < 3; ++it) {
    hipMemsetAsync(emaxk, 0, (size_t)GG*4, stream);
    hipMemsetAsync(asum,  0, (size_t)GG*4, stream);
    hipMemsetAsync(rvec,  0, (size_t)GG*DIM*4, stream);
    k_lstm  <<<GG/8,256,0,stream>>>(qstar, qh, qc, lstm_wih, lstm_whh, lstm_bih, lstm_bhh);
    k_escore<<<NN/256,256,0,stream>>>(out_s, qh, node_batch, escore, emaxk);
    k_aexp  <<<NN/256,256,0,stream>>>(escore, emaxk, node_batch, avec, asum);
    k_rvec  <<<NN/8,256,0,stream>>>(out_s, avec, asum, node_batch, rvec);
    k_qstar <<<GG*128/256,256,0,stream>>>(qh, rvec, qstar);
  }

  // predict head: gather -> WMMA MLP x2 -> final projection + class gather
  k_gatherP<<<TT/8,256,0,stream>>>(out_s, qstar, target_index, target_batch, P);
  k_mlp<<<TT/16,256,0,stream>>>(P, w1pbf, prd_b1, prd_g1, prd_be1, Q);
  k_mlp<<<TT/16,256,0,stream>>>(Q, w2pbf, prd_b2, prd_g2, prd_be2, P);
  k_final<<<TT/8,256,0,stream>>>(P, prd_w3, prd_b3, target_class, (float*)d_out);
}
